// kernel_generated_0_62904091017250
// MI455X (gfx1250) — matrix operand order repaired and checked by reading, not compiled, not run on hardware
//
#include <hip/hip_runtime.h>

// MI455X / gfx1250, wave32. Fused: t4 GEMM (M=56,N=64,K=192) via
// v_wmma_f32_16x16x32_bf16 with 2-term bf16 split (hh+hl+lh), plus the
// circular-shift w0 mix done from registers (cur/prev slice accumulators).

typedef __attribute__((ext_vector_type(16))) __bf16 v16bf;
typedef __attribute__((ext_vector_type(8)))  __bf16 v8bf;
typedef __attribute__((ext_vector_type(8)))  float  v8f;

namespace {

__device__ __forceinline__ __bf16 f2bf(float f) {
  unsigned u = __builtin_bit_cast(unsigned, f);
  unsigned short h = (unsigned short)((u + 0x7FFFu + ((u >> 16) & 1u)) >> 16);
  return __builtin_bit_cast(__bf16, h);
}
__device__ __forceinline__ float bf2f(__bf16 b) {
  unsigned short h = __builtin_bit_cast(unsigned short, b);
  unsigned u = ((unsigned)h) << 16;
  return __builtin_bit_cast(float, u);
}

#define XROWS 66   // rows h+1 in [0,65]; rows 0 and 57..65 are static zeros

__global__ __launch_bounds__(256) void fused_unfold_mix(
    const float* __restrict__ x,    // (1024, 64, 56, 56) viewed from (1024,3584,56)
    const float* __restrict__ w0,   // (64, 2)
    const float* __restrict__ w1,   // (64, 3, 64)
    float* __restrict__ out) {      // (1024, 64, 56, 56)
  __shared__ __align__(16) __bf16 Xhi[XROWS * 64];
  __shared__ __align__(16) __bf16 Xlo[XROWS * 64];
  __shared__ __align__(16) __bf16 BloLds[6 * 4 * 32 * 16]; // [c][nt][lane][e]
  __shared__ float Obuf[64 * 57];                          // pitch 57: conflict-free

  const int tid   = threadIdx.x;
  const int lane  = tid & 31;
  const int wv    = tid >> 5;
  const int b     = blockIdx.x;
  const int mt    = wv & 3;            // M tile (n)
  const int nt0   = (wv >> 2) * 2;     // N tiles (i)
  const int nt1   = nt0 + 1;
  const int lHalf = lane >> 4;
  const int lLow  = lane & 15;

  // ---- B-hi fragments (WMMA 16-bit B 32x16 layout) held in registers ----
  // K index kk = k*64 + j ; chunk c: k = c>>1, j = (c&1)*32 + (lane<16?0:16) + e
  v16bf bhi[6][2];
  {
    const int kb = lHalf * 16;
#pragma unroll
    for (int c = 0; c < 6; ++c) {
      const int k  = c >> 1;
      const int jb = (c & 1) * 32 + kb;
#pragma unroll
      for (int t = 0; t < 2; ++t) {
        const int i = (nt0 + t) * 16 + lLow;
        union { __bf16 h[16]; v16bf v; } u;
#pragma unroll
        for (int e = 0; e < 16; ++e)
          u.h[e] = f2bf(w1[((jb + e) * 3 + k) * 64 + i]);
        bhi[c][t] = u.v;
      }
    }
  }
  // ---- B-lo fragments into LDS (all 4 nt, all lanes) ----
  for (int idx = tid; idx < 6 * 4 * 32 * 16; idx += 256) {
    const int e  = idx & 15;
    const int ln = (idx >> 4) & 31;
    const int nt = (idx >> 9) & 3;
    const int c  = idx >> 11;
    const int j  = (c & 1) * 32 + ((ln < 16) ? 0 : 16) + e;
    const float f = w1[(j * 3 + (c >> 1)) * 64 + nt * 16 + (ln & 15)];
    const __bf16 hi = f2bf(f);
    BloLds[idx] = f2bf(f - bf2f(hi));
  }
  // ---- zero pad rows of Xhi/Xlo (rows 0 and 57..65) ----
  for (int idx = tid; idx < 10 * 64; idx += 256) {
    const int rr  = idx >> 6;
    const int row = (rr == 0) ? 0 : (56 + rr);
    Xhi[row * 64 + (idx & 63)] = f2bf(0.0f);
    Xlo[row * 64 + (idx & 63)] = f2bf(0.0f);
  }

  // per-lane w0 scalars: cur slice pairs w0[:,0], prev slice pairs w0[:,1]
  const int iC0 = nt0 * 16 + lLow;
  const int iC1 = nt1 * 16 + lLow;
  const float w0c0 = w0[iC0 * 2 + 0], w0p0 = w0[iC0 * 2 + 1];
  const float w0c1 = w0[iC1 * 2 + 0], w0p1 = w0[iC1 * 2 + 1];

  const int jj = tid >> 2;   // staging: j channel
  const int q  = tid & 3;    // staging: quarter of the h row

  v8f prev0 = {}, prev1 = {};

  for (int ll = 0; ll < 57; ++ll) {
    const int l = (ll == 0) ? 55 : (ll - 1);
    __syncthreads();
    // ---- stage x[b, :, l, :] -> Xhi/Xlo[h+1][j] (bf16 split) ----
    {
      const float* src = x + (((size_t)b * 64 + jj) * 56 + l) * 56;
#pragma unroll
      for (int f4i = 0; f4i < 4; ++f4i) {
        const int f4 = q + 4 * f4i;          // float4 index 0..13
        if (f4 < 14) {
          const float4 v = *(const float4*)(src + f4 * 4);
          const float vs[4] = {v.x, v.y, v.z, v.w};
#pragma unroll
          for (int d = 0; d < 4; ++d) {
            const int row = f4 * 4 + d + 1;  // h+1
            const __bf16 hi = f2bf(vs[d]);
            Xhi[row * 64 + jj] = hi;
            Xlo[row * 64 + jj] = f2bf(vs[d] - bf2f(hi));
          }
        }
      }
    }
    __syncthreads();

    // ---- GEMM for this slice: acc = Ahi*Bhi + Ahi*Blo + Alo*Bhi ----
    v8f acc0 = {}, acc1 = {};
    const int rowb = mt * 16 + lLow;         // n
    const int j0b  = lHalf * 16;             // A-layout lane-half K offset
#pragma unroll
    for (int c = 0; c < 6; ++c) {
      const int row = rowb + (c >> 1);       // (h+1) = n + k
      const int off = row * 64 + (c & 1) * 32 + j0b;
      v8bf alo8, ahi8l, ahi8h, alo8h;
      ahi8l = *(const v8bf*)&Xhi[off];
      ahi8h = *(const v8bf*)&Xhi[off + 8];
      alo8  = *(const v8bf*)&Xlo[off];
      alo8h = *(const v8bf*)&Xlo[off + 8];
      v16bf ahi = __builtin_shufflevector(ahi8l, ahi8h, 0,1,2,3,4,5,6,7,8,9,10,11,12,13,14,15);
      v16bf alo = __builtin_shufflevector(alo8,  alo8h, 0,1,2,3,4,5,6,7,8,9,10,11,12,13,14,15);
      const v8bf* bl0 = (const v8bf*)&BloLds[(((c * 4 + nt0) * 32) + lane) * 16];
      const v8bf* bl1 = (const v8bf*)&BloLds[(((c * 4 + nt1) * 32) + lane) * 16];
      v16bf blo0 = __builtin_shufflevector(bl0[0], bl0[1], 0,1,2,3,4,5,6,7,8,9,10,11,12,13,14,15);
      v16bf blo1 = __builtin_shufflevector(bl1[0], bl1[1], 0,1,2,3,4,5,6,7,8,9,10,11,12,13,14,15);
      acc0 = __builtin_amdgcn_wmma_f32_16x16x32_bf16(false, ahi, false, bhi[c][0], (short)0, acc0, false, false);
      acc1 = __builtin_amdgcn_wmma_f32_16x16x32_bf16(false, ahi, false, bhi[c][1], (short)0, acc1, false, false);
      acc0 = __builtin_amdgcn_wmma_f32_16x16x32_bf16(false, ahi, false, blo0,      (short)0, acc0, false, false);
      acc1 = __builtin_amdgcn_wmma_f32_16x16x32_bf16(false, ahi, false, blo1,      (short)0, acc1, false, false);
      acc0 = __builtin_amdgcn_wmma_f32_16x16x32_bf16(false, alo, false, bhi[c][0], (short)0, acc0, false, false);
      acc1 = __builtin_amdgcn_wmma_f32_16x16x32_bf16(false, alo, false, bhi[c][1], (short)0, acc1, false, false);
    }

    if (ll > 0) {
      const int k2 = ll % 56;  // out[k2] = t4[k2-1]*w0[:,0] + t4[k2-2]*w0[:,1]
#pragma unroll
      for (int r = 0; r < 8; ++r) {
        const int n = mt * 16 + r + lHalf * 8;   // C/D layout: M = r (+8 upper half)
        if (n < 56) {
          Obuf[iC0 * 57 + n] = fmaf(acc0[r], w0c0, prev0[r] * w0p0);
          Obuf[iC1 * 57 + n] = fmaf(acc1[r], w0c1, prev1[r] * w0p1);
        }
      }
      __syncthreads();
      float* obase = out + (size_t)b * (64 * 56 * 56) + (size_t)k2 * 56;
      for (int idx = tid; idx < 64 * 56; idx += 256) {
        const int i = idx / 56;
        const int m = idx - i * 56;
        obase[(size_t)i * (56 * 56) + m] = Obuf[i * 57 + m];
      }
    }
    prev0 = acc0;
    prev1 = acc1;
  }
}

} // namespace

extern "C" void kernel_launch(void* const* d_in, const int* in_sizes, int n_in,
                              void* d_out, int out_size, void* d_ws, size_t ws_size,
                              hipStream_t stream) {
  (void)in_sizes; (void)n_in; (void)out_size; (void)d_ws; (void)ws_size;
  const float* x  = (const float*)d_in[0];
  const float* w0 = (const float*)d_in[1];
  const float* w1 = (const float*)d_in[2];
  float* out = (float*)d_out;
  fused_unfold_mix<<<dim3(1024), dim3(256), 0, stream>>>(x, w0, w1, out);
}
